// MouseSRNN_14396730376242
// MI455X (gfx1250) — matrix operand order repaired and checked by reading, not compiled, not run on hardware
//
#include <hip/hip_runtime.h>
#include <hip/hip_bf16.h>
#include <math.h>

// ---------------------------------------------------------------------------
// MouseSRNN on MI455X (gfx1250): f16 WMMA (f32 accum) recurrent network.
// - v_wmma_f32_16x16x32_f16 GEMMs, weights pre-packed fragment-linear
// - X tiles staged to LDS via global_load_async_to_lds_b128 (ASYNCcnt)
// - fused gates GEMM + LSTM nonlinearity in registers (no gates round-trip)
// ---------------------------------------------------------------------------

typedef _Float16 half_t;
typedef __attribute__((ext_vector_type(16))) _Float16 v16h;
typedef __attribute__((ext_vector_type(8)))  _Float16 v8h;
typedef __attribute__((ext_vector_type(8)))  float    v8f;

#define B_    64
#define T_    32
#define N_    12
#define NS_   132
#define ATTN_ 64
#define KN_   11          // NS/N neighbors per node
#define MSPAT (B_*NS_)    // 8448 rows, 528 m-tiles
#define MNODE (B_*N_)     // 768 rows,  48 m-tiles
#define TEMP_ 1.375f      // KN_/sqrt(ATTN_)

__device__ __forceinline__ float sigmoidf_(float x) { return 1.0f / (1.0f + __expf(-x)); }
__device__ __forceinline__ float tanh_fast(float x) {
  x = fminf(fmaxf(x, -15.0f), 15.0f);
  float e = __expf(2.0f * x);
  return (e - 1.0f) / (e + 1.0f);
}

__device__ __forceinline__ v16h make_frag(v8h lo, v8h hi) {
  v16h r;
#pragma unroll
  for (int i = 0; i < 8; ++i) { r[i] = lo[i]; r[8 + i] = hi[i]; }
  return r;
}

// A fragment (16x32 f16, MxK) from LDS: lane m(0..15) holds row m; lanes 0-15
// K={k0..k0+7, k0+16..k0+23}, lanes 16-31 K={k0+8..k0+15, k0+24..k0+31}.
__device__ __forceinline__ v16h load_a_frag(const half_t* xrow, int k0, int khalf) {
  const v8h* p0 = (const v8h*)(xrow + k0 + khalf * 8);
  const v8h* p1 = (const v8h*)(xrow + k0 + 16 + khalf * 8);
  return make_frag(*p0, *p1);
}

// B fragment from fragment-linear packed weights: frag f is 512 contiguous
// halves; lane l owns halves [l*16, l*16+16) -> fully coalesced 1KB per wave.
__device__ __forceinline__ v16h load_b_frag_packed(const half_t* __restrict__ Wp,
                                                   int f, int lane) {
  const v8h* p = (const v8h*)(Wp + ((size_t)f << 9) + lane * 16);
  return make_frag(p[0], p[1]);
}

#define WMMA_F16(a, b, c) \
  __builtin_amdgcn_wmma_f32_16x16x32_f16(false, (a), false, (b), (short)0, (c), false, false)

// Stage 16 rows x K halves of [A1|A2] into LDS with async direct-to-LDS copies
// (GLOBAL_LOAD_ASYNC_TO_LDS_B128, tracked by ASYNCcnt; no VGPR round-trip).
__device__ __forceinline__ void stage_x_async(half_t* xtile,
                                              const half_t* __restrict__ A1, int ld1, int k1,
                                              const half_t* __restrict__ A2, int ld2,
                                              int K, int m0, int tid) {
  const int kc8 = K >> 3;        // 16-byte chunks per row
  const int chunks = 16 * kc8;   // total chunks for the 16-row tile
  for (int c = tid; c < chunks; c += 256) {
    int row_local = c / kc8;
    int kc = (c - row_local * kc8) << 3;
    int row = m0 + row_local;
    const half_t* src = (kc < k1) ? (A1 + (size_t)row * ld1 + kc)
                                  : (A2 + (size_t)row * ld2 + (kc - k1));
    // LDS generic addresses carry the LDS byte offset in the low 32 bits.
    unsigned lds_off = (unsigned)(size_t)(xtile + row_local * K + kc);
    asm volatile("global_load_async_to_lds_b128 %0, %1, off"
                 :: "v"(lds_off), "v"(src) : "memory");
  }
  asm volatile("s_wait_asynccnt 0x0" ::: "memory");
}

// ---------------------------------------------------------------------------
// Weight prep: f32 -> f16, concat [Wih|Whh], permuted into fragment-linear
// layout. For matrix [Nout x K]: frag f = nt*(K/32)+ks holds, at lane l,
// element e: W[nt*16+(l&15)][ks*32+(l>>4)*8+(e&7)+(e>>3)*16].
// ---------------------------------------------------------------------------
__device__ __forceinline__ void unpack_idx(int i, int K, int& n, int& k) {
  int f = i >> 9, r = i & 511;
  int lane = r >> 4, e = r & 15;
  int kcnt = K >> 5;
  int nt = f / kcnt, ks = f - nt * kcnt;
  n = nt * 16 + (lane & 15);
  k = ks * 32 + (lane >> 4) * 8 + (e & 7) + (e >> 3) * 16;
}

__global__ __launch_bounds__(256) void prep_kernel(
    const float* __restrict__ se_Wih, const float* __restrict__ se_Whh,
    const float* __restrict__ te_Wih, const float* __restrict__ te_Whh,
    const float* __restrict__ nr_Wih, const float* __restrict__ nr_Whh,
    const float* __restrict__ W_edge, const float* __restrict__ Wq,
    const float* __restrict__ Wk,
    const float* __restrict__ se_bih, const float* __restrict__ se_bhh,
    const float* __restrict__ te_bih, const float* __restrict__ te_bhh,
    const float* __restrict__ nr_bih, const float* __restrict__ nr_bhh,
    half_t* __restrict__ Wspat16, half_t* __restrict__ Wtemp16,
    half_t* __restrict__ Wnode16, half_t* __restrict__ Wedge16,
    half_t* __restrict__ Wq16, half_t* __restrict__ Wk16,
    float* __restrict__ bspat, float* __restrict__ btemp, float* __restrict__ bnode) {
  int i = blockIdx.x * 256 + threadIdx.x;
  int n, k;
  if (i < 393216) { // Wspat [1024 x 384]
    unpack_idx(i, 384, n, k);
    Wspat16[i] = (half_t)((k < 128) ? se_Wih[n * 128 + k] : se_Whh[n * 256 + k - 128]);
    return;
  }
  i -= 393216;
  if (i < 393216) { // Wtemp [1024 x 384]
    unpack_idx(i, 384, n, k);
    Wtemp16[i] = (half_t)((k < 128) ? te_Wih[n * 128 + k] : te_Whh[n * 256 + k - 128]);
    return;
  }
  i -= 393216;
  if (i < 524288) { // Wnode [1024 x 512]
    unpack_idx(i, 512, n, k);
    Wnode16[i] = (half_t)((k < 256) ? nr_Wih[n * 256 + k] : nr_Whh[n * 256 + k - 256]);
    return;
  }
  i -= 524288;
  if (i < 65536) { // Wedge [128 x 512]
    unpack_idx(i, 512, n, k);
    Wedge16[i] = (half_t)W_edge[n * 512 + k];
    return;
  }
  i -= 65536;
  if (i < 16384) { // Wq [64 x 256]
    unpack_idx(i, 256, n, k);
    Wq16[i] = (half_t)Wq[n * 256 + k];
    return;
  }
  i -= 16384;
  if (i < 16384) { // Wk [64 x 256]
    unpack_idx(i, 256, n, k);
    Wk16[i] = (half_t)Wk[n * 256 + k];
    return;
  }
  i -= 16384;
  if (i < 1024) { bspat[i] = se_bih[i] + se_bhh[i]; return; }
  i -= 1024;
  if (i < 1024) { btemp[i] = te_bih[i] + te_bhh[i]; return; }
  i -= 1024;
  if (i < 1024) { bnode[i] = nr_bih[i] + nr_bhh[i]; return; }
}

// ---------------------------------------------------------------------------
// Tiny K=2 input embeddings (se / te / nin), fused into one launch.
// ---------------------------------------------------------------------------
__global__ __launch_bounds__(256) void embed_kernel(
    const float* __restrict__ nodes, const float* __restrict__ et,
    const float* __restrict__ es,
    const float* __restrict__ W_te, const float* __restrict__ b_te,
    const float* __restrict__ W_se, const float* __restrict__ b_se,
    const float* __restrict__ W_node, const float* __restrict__ b_node,
    half_t* __restrict__ se16, half_t* __restrict__ te16,
    half_t* __restrict__ rin16, int t) {
  const int SE_TOT = MSPAT * 128;
  const int TE_TOT = MNODE * 128;
  int i = blockIdx.x * 256 + threadIdx.x;
  if (i < SE_TOT) {
    int j = i & 127, row = i >> 7;
    int s = row % NS_, b = row / NS_;
    const float* in = es + (((size_t)b * T_ + t) * NS_ + s) * 2;
    float v = fmaf(W_se[2 * j], in[0], fmaf(W_se[2 * j + 1], in[1], b_se[j]));
    se16[(size_t)row * 128 + j] = (half_t)fmaxf(v, 0.0f);
    return;
  }
  i -= SE_TOT;
  if (i < TE_TOT) {
    int j = i & 127, row = i >> 7;
    int n = row % N_, b = row / N_;
    const float* in = et + (((size_t)b * T_ + t) * N_ + n) * 2;
    float v = fmaf(W_te[2 * j], in[0], fmaf(W_te[2 * j + 1], in[1], b_te[j]));
    te16[(size_t)row * 128 + j] = (half_t)fmaxf(v, 0.0f);
    return;
  }
  i -= TE_TOT;
  if (i < TE_TOT) {
    int j = i & 127, row = i >> 7;
    int n = row % N_, b = row / N_;
    const float* in = nodes + (((size_t)b * T_ + t) * N_ + n) * 2;
    float v = fmaf(W_node[2 * j], in[0], fmaf(W_node[2 * j + 1], in[1], b_node[j]));
    rin16[(size_t)row * 256 + j] = (half_t)fmaxf(v, 0.0f); // nin -> rin cols 0..127
  }
}

// ---------------------------------------------------------------------------
// Fused LSTM step: gates = [X1|X2] @ Wcat^T (+bsum) via WMMA, then the
// sigmoid/tanh update runs directly on the accumulators. Wave w owns hidden
// column tiles {w, w+8} across all four gate blocks -> i/f/g/o co-resident.
// ---------------------------------------------------------------------------
__global__ __launch_bounds__(256) void lstm_step_kernel(
    const half_t* __restrict__ A1, int ld1, int k1,
    const half_t* __restrict__ A2, int ld2,
    const half_t* __restrict__ W, const float* __restrict__ bsum, int K,
    float* __restrict__ c_state, half_t* __restrict__ h16) {
  __shared__ alignas(16) half_t xtile[16 * 512];
  const int tid  = threadIdx.x;
  const int wave = tid >> 5;
  const int lane = tid & 31;
  const int m0   = blockIdx.x * 16;
  const int kcnt = K >> 5;

  stage_x_async(xtile, A1, ld1, k1, A2, ld2, K, m0, tid);
  __syncthreads();

  const half_t* xrow = xtile + (size_t)(lane & 15) * K;
  const int khalf = lane >> 4;

  v8f acc[2][4];
#pragma unroll
  for (int g = 0; g < 2; ++g)
#pragma unroll
    for (int gate = 0; gate < 4; ++gate)
#pragma unroll
      for (int r = 0; r < 8; ++r) acc[g][gate][r] = 0.0f;

  for (int ks = 0; ks < kcnt; ++ks) {
    v16h a = load_a_frag(xrow, ks << 5, khalf);
#pragma unroll
    for (int g = 0; g < 2; ++g) {
      int ntbase = wave + 8 * g;
#pragma unroll
      for (int gate = 0; gate < 4; ++gate) {
        v16h b = load_b_frag_packed(W, (gate * 16 + ntbase) * kcnt + ks, lane);
        acc[g][gate] = WMMA_F16(a, b, acc[g][gate]);
      }
    }
  }

  // c2 = sig(f)*c + sig(i)*tanh(g); h2 = sig(o)*tanh(c2)
#pragma unroll
  for (int g = 0; g < 2; ++g) {
    int col = (wave + 8 * g) * 16 + (lane & 15);
    float bi = bsum[col], bf = bsum[256 + col], bg = bsum[512 + col], bo = bsum[768 + col];
#pragma unroll
    for (int r = 0; r < 8; ++r) {
      int row = m0 + ((lane < 16) ? r : r + 8);
      size_t idx = (size_t)row * 256 + col;
      float iv = acc[g][0][r] + bi;
      float fv = acc[g][1][r] + bf;
      float gv = acc[g][2][r] + bg;
      float ov = acc[g][3][r] + bo;
      float c2 = sigmoidf_(fv) * c_state[idx] + sigmoidf_(iv) * tanh_fast(gv);
      float h2 = sigmoidf_(ov) * tanh_fast(c2);
      c_state[idx] = c2;
      h16[idx] = (half_t)h2;
    }
  }
}

// ---------------------------------------------------------------------------
// Generic WMMA GEMM: out = [X1|X2] @ W^T + bias, optional relu, f32/f16 out.
// ---------------------------------------------------------------------------
__global__ __launch_bounds__(256) void wmma_gemm_kernel(
    const half_t* __restrict__ A1, int ld1, int k1,
    const half_t* __restrict__ A2, int ld2,
    const half_t* __restrict__ W, const float* __restrict__ bias,
    int K, int Ntiles,
    float* __restrict__ out32, half_t* __restrict__ out16,
    int out_ld, int out_coloff, int do_relu) {
  __shared__ alignas(16) half_t xtile[16 * 512];
  const int tid  = threadIdx.x;
  const int wave = tid >> 5;
  const int lane = tid & 31;
  const int m0   = blockIdx.x * 16;
  const int kcnt = K >> 5;

  stage_x_async(xtile, A1, ld1, k1, A2, ld2, K, m0, tid);
  __syncthreads();

  if (wave < Ntiles) {
    const half_t* xrow = xtile + (size_t)(lane & 15) * K;
    const int khalf = lane >> 4;
    v8f acc;
#pragma unroll
    for (int r = 0; r < 8; ++r) acc[r] = 0.0f;
    for (int ks = 0; ks < kcnt; ++ks) {
      v16h a = load_a_frag(xrow, ks << 5, khalf);
      v16h b = load_b_frag_packed(W, wave * kcnt + ks, lane);
      acc = WMMA_F16(a, b, acc);
    }
    int nloc = wave * 16 + (lane & 15);
    int col = out_coloff + nloc;
    float bv = bias[nloc];
#pragma unroll
    for (int r = 0; r < 8; ++r) {
      int row = m0 + ((lane < 16) ? r : r + 8);
      float v = acc[r] + bv;
      if (do_relu) v = fmaxf(v, 0.0f);
      if (out32) out32[(size_t)row * out_ld + col] = v;
      if (out16) out16[(size_t)row * out_ld + col] = (half_t)v;
    }
  }
}

// ---------------------------------------------------------------------------
// Attention over the 11 neighbor edges of each node (wave per node).
// ---------------------------------------------------------------------------
__global__ __launch_bounds__(256) void attn_kernel(
    const float* __restrict__ qbuf, const float* __restrict__ kbuf,
    const half_t* __restrict__ h_spat16, half_t* __restrict__ h_attn16) {
  int node = blockIdx.x * 8 + (threadIdx.x >> 5); // 0..767
  int lane = threadIdx.x & 31;
  int b = node / N_, n = node % N_;
  int erow0 = b * NS_ + n * KN_;     // NODE_TO_EDGES = arange: contiguous rows
  const float* qrow = qbuf + (size_t)node * ATTN_;

  float s;
  if (lane < KN_) {
    const float* krow = kbuf + (size_t)(erow0 + lane) * ATTN_;
    float acc = 0.0f;
    for (int j = 0; j < ATTN_; ++j) acc = fmaf(qrow[j], krow[j], acc);
    s = acc * TEMP_;
  } else {
    s = -3.0e38f;
  }
  float m = s;
#pragma unroll
  for (int off = 16; off > 0; off >>= 1) m = fmaxf(m, __shfl_xor(m, off, 32));
  float e = (lane < KN_) ? __expf(s - m) : 0.0f;
  float sum = e;
#pragma unroll
  for (int off = 16; off > 0; off >>= 1) sum += __shfl_xor(sum, off, 32);
  float w = e / sum;

  float accv[8];
#pragma unroll
  for (int j = 0; j < 8; ++j) accv[j] = 0.0f;
  for (int kk = 0; kk < KN_; ++kk) {
    float wk = __shfl(w, kk, 32);
    const half_t* hrow = h_spat16 + (size_t)(erow0 + kk) * 256 + lane * 8;
#pragma unroll
    for (int j = 0; j < 8; ++j) accv[j] = fmaf(wk, (float)hrow[j], accv[j]);
  }
  half_t* outp = h_attn16 + (size_t)node * 256 + lane * 8;
#pragma unroll
  for (int j = 0; j < 8; ++j) outp[j] = (half_t)accv[j];
}

// ---------------------------------------------------------------------------
// Output head: (768 x 256) @ (256 x 5), scattered into (B,T,N,5).
// ---------------------------------------------------------------------------
__global__ __launch_bounds__(256) void out_kernel(
    const half_t* __restrict__ h_node16, const float* __restrict__ W_out,
    const float* __restrict__ b_out, float* __restrict__ out, int t) {
  int gid = blockIdx.x * 256 + threadIdx.x;
  if (gid >= MNODE * 5) return;
  int row = gid / 5, o = gid % 5;
  int b = row / N_, n = row % N_;
  const half_t* h = h_node16 + (size_t)row * 256;
  const float* w = W_out + o * 256;
  float s = b_out[o];
  for (int j = 0; j < 256; ++j) s = fmaf((float)h[j], w[j], s);
  out[(((size_t)b * T_ + t) * N_ + n) * 5 + o] = s;
}

// ---------------------------------------------------------------------------
extern "C" void kernel_launch(void* const* d_in, const int* in_sizes, int n_in,
                              void* d_out, int out_size, void* d_ws, size_t ws_size,
                              hipStream_t stream) {
  (void)in_sizes; (void)n_in; (void)out_size; (void)ws_size;
  const float* nodes   = (const float*)d_in[0];
  const float* et      = (const float*)d_in[1];
  const float* es      = (const float*)d_in[2];
  const float* W_te    = (const float*)d_in[3];
  const float* b_te    = (const float*)d_in[4];
  const float* te_bih  = (const float*)d_in[7];
  const float* te_bhh  = (const float*)d_in[8];
  const float* W_se    = (const float*)d_in[9];
  const float* b_se    = (const float*)d_in[10];
  const float* se_Wih  = (const float*)d_in[11];
  const float* se_Whh  = (const float*)d_in[12];
  const float* se_bih  = (const float*)d_in[13];
  const float* se_bhh  = (const float*)d_in[14];
  const float* te_Wih  = (const float*)d_in[5];
  const float* te_Whh  = (const float*)d_in[6];
  const float* Wq      = (const float*)d_in[15];
  const float* bq      = (const float*)d_in[16];
  const float* Wk      = (const float*)d_in[17];
  const float* bk      = (const float*)d_in[18];
  const float* W_node  = (const float*)d_in[19];
  const float* b_node  = (const float*)d_in[20];
  const float* W_edge  = (const float*)d_in[21];
  const float* b_edge  = (const float*)d_in[22];
  const float* nr_Wih  = (const float*)d_in[23];
  const float* nr_Whh  = (const float*)d_in[24];
  const float* nr_bih  = (const float*)d_in[25];
  const float* nr_bhh  = (const float*)d_in[26];
  const float* W_out   = (const float*)d_in[27];
  const float* b_out   = (const float*)d_in[28];
  float* out = (float*)d_out;

  // --- workspace carve-up (256B aligned slices) ---
  char* base = (char*)d_ws;
  size_t off = 0;
  auto alloc = [&](size_t bytes) -> void* {
    void* r = base + off;
    off += (bytes + 255) & ~(size_t)255;
    return r;
  };
  half_t* Wspat16  = (half_t*)alloc((size_t)1024 * 384 * 2);
  half_t* Wtemp16  = (half_t*)alloc((size_t)1024 * 384 * 2);
  half_t* Wnode16  = (half_t*)alloc((size_t)1024 * 512 * 2);
  half_t* Wedge16  = (half_t*)alloc((size_t)128 * 512 * 2);
  half_t* Wq16     = (half_t*)alloc((size_t)64 * 256 * 2);
  half_t* Wk16     = (half_t*)alloc((size_t)64 * 256 * 2);
  float*  bspat    = (float*)alloc(1024 * 4);
  float*  btemp    = (float*)alloc(1024 * 4);
  float*  bnode    = (float*)alloc(1024 * 4);
  half_t* se16     = (half_t*)alloc((size_t)MSPAT * 128 * 2);
  half_t* te16     = (half_t*)alloc((size_t)MNODE * 128 * 2);
  half_t* rin16    = (half_t*)alloc((size_t)MNODE * 256 * 2);
  half_t* h_spat16 = (half_t*)alloc((size_t)MSPAT * 256 * 2);
  float*  c_spat   = (float*)alloc((size_t)MSPAT * 256 * 4);
  half_t* h_temp16 = (half_t*)alloc((size_t)MNODE * 256 * 2);
  float*  c_temp   = (float*)alloc((size_t)MNODE * 256 * 4);
  half_t* h_node16 = (half_t*)alloc((size_t)MNODE * 256 * 2);
  float*  c_node   = (float*)alloc((size_t)MNODE * 256 * 4);
  float*  qbuf     = (float*)alloc((size_t)MNODE * 64 * 4);
  float*  kbuf     = (float*)alloc((size_t)MSPAT * 64 * 4);
  half_t* h_attn16 = (half_t*)alloc((size_t)MNODE * 256 * 2);

  // zero recurrent state every call (deterministic)
  hipMemsetAsync(h_spat16, 0, (size_t)MSPAT * 256 * 2, stream);
  hipMemsetAsync(c_spat,   0, (size_t)MSPAT * 256 * 4, stream);
  hipMemsetAsync(h_temp16, 0, (size_t)MNODE * 256 * 2, stream);
  hipMemsetAsync(c_temp,   0, (size_t)MNODE * 256 * 4, stream);
  hipMemsetAsync(h_node16, 0, (size_t)MNODE * 256 * 2, stream);
  hipMemsetAsync(c_node,   0, (size_t)MNODE * 256 * 4, stream);

  // weight prep (1,412,096 elements)
  prep_kernel<<<(1412096 + 255) / 256, 256, 0, stream>>>(
      se_Wih, se_Whh, te_Wih, te_Whh, nr_Wih, nr_Whh, W_edge, Wq, Wk,
      se_bih, se_bhh, te_bih, te_bhh, nr_bih, nr_bhh,
      Wspat16, Wtemp16, Wnode16, Wedge16, Wq16, Wk16, bspat, btemp, bnode);

  const int EMBED_TOT = MSPAT * 128 + 2 * MNODE * 128; // 1,277,952
  for (int t = 0; t < T_; ++t) {
    // 1) input embeddings (K=2 projections, VALU)
    embed_kernel<<<(EMBED_TOT + 255) / 256, 256, 0, stream>>>(
        nodes, et, es, W_te, b_te, W_se, b_se, W_node, b_node,
        se16, te16, rin16, t);
    // 2) spatial LSTM: X=[se16|h_spat16], K=384, M=8448
    lstm_step_kernel<<<MSPAT / 16, 256, 0, stream>>>(
        se16, 128, 128, h_spat16, 256, Wspat16, bspat, 384, c_spat, h_spat16);
    // 3) temporal LSTM: X=[te16|h_temp16], K=384, M=768
    lstm_step_kernel<<<MNODE / 16, 256, 0, stream>>>(
        te16, 128, 128, h_temp16, 256, Wtemp16, btemp, 384, c_temp, h_temp16);
    // 4) q = h_temp @ Wq^T + bq ; k = h_spat @ Wk^T + bk
    wmma_gemm_kernel<<<MNODE / 16, 256, 0, stream>>>(
        h_temp16, 256, 256, h_temp16, 256, Wq16, bq, 256, 4,
        qbuf, (half_t*)nullptr, 64, 0, 0);
    wmma_gemm_kernel<<<MSPAT / 16, 256, 0, stream>>>(
        h_spat16, 256, 256, h_spat16, 256, Wk16, bk, 256, 4,
        kbuf, (half_t*)nullptr, 64, 0, 0);
    // 5) neighbor attention -> h_attn16
    attn_kernel<<<MNODE / 8, 256, 0, stream>>>(qbuf, kbuf, h_spat16, h_attn16);
    // 6) ein = relu([h_temp|h_attn] @ W_edge^T + b_edge) -> rin cols 128..255
    wmma_gemm_kernel<<<MNODE / 16, 256, 0, stream>>>(
        h_temp16, 256, 256, h_attn16, 256, Wedge16, b_edge, 512, 8,
        (float*)nullptr, rin16, 256, 128, 1);
    // 7) node LSTM: X=[rin16|h_node16], K=512, M=768
    lstm_step_kernel<<<MNODE / 16, 256, 0, stream>>>(
        rin16, 256, 256, h_node16, 256, Wnode16, bnode, 512, c_node, h_node16);
    // 8) output head
    out_kernel<<<(MNODE * 5 + 255) / 256, 256, 0, stream>>>(
        h_node16, W_out, b_out, out, t);
  }
}
